// E2E_19602230739944
// MI455X (gfx1250) — compile-verified, matrix operand order checked by reading, not run on hardware
//
#include <hip/hip_runtime.h>

// ---- problem constants (match reference) ----
#define B_   32
#define T_   300
#define ENC_ 1024
#define H_   512
#define A_   256
#define C_   10
#define KK_  100      // conv half width
#define V_   10000
#define E_   512
#define AO_  512
#define TD_  80
#define KX_  1536     // E + AO + H  (lstm input concat [emb | c | dec_z])
#define G4_  2048     // 4*H
#define KO_  1024     // H + AO      (logit input concat [dec_z | c])

typedef __attribute__((ext_vector_type(16))) __bf16 v16bf;
typedef __attribute__((ext_vector_type(8)))  float  v8f;

union BfVec { v16bf v; unsigned int u[8]; };

static __device__ __forceinline__ unsigned short f2bf(float f) {
    unsigned int u = __float_as_uint(f);
    u += 0x7fffu + ((u >> 16) & 1u);      // round to nearest even
    return (unsigned short)(u >> 16);
}
static __device__ __forceinline__ float sigmf(float x) {
    return 1.0f / (1.0f + __expf(-x));
}

// ---------------- setup kernels ----------------

// Wcat[n, 0:1024] = W_ih[n,:]; Wcat[n, 1024:1536] = W_hh[n,:]   (bf16)
__global__ void k_cvt_wcat(const float* __restrict__ Wih, const float* __restrict__ Whh,
                           unsigned short* __restrict__ Wcat) {
    size_t i = (size_t)blockIdx.x * blockDim.x + threadIdx.x;
    if (i >= (size_t)G4_ * KX_) return;
    int n = (int)(i / KX_), k = (int)(i % KX_);
    float v = (k < 1024) ? Wih[(size_t)n * 1024 + k] : Whh[(size_t)n * 512 + (k - 1024)];
    Wcat[i] = f2bf(v);
}

__global__ void k_cvt_wout(const float* __restrict__ W, unsigned short* __restrict__ Wb) {
    size_t i = (size_t)blockIdx.x * blockDim.x + threadIdx.x;
    if (i >= (size_t)V_ * KO_) return;
    Wb[i] = f2bf(W[i]);
}

__global__ void k_bias(const float* __restrict__ bih, const float* __restrict__ bhh,
                       float* __restrict__ gb) {
    int i = blockIdx.x * blockDim.x + threadIdx.x;
    if (i < G4_) gb[i] = bih[i] + bhh[i];
}

// pre_enc[b,t,a] = sum_d enc[b,t,d]*W_enc[a,d] + b_enc[a]
__global__ void k_preenc(const float* __restrict__ enc, const float* __restrict__ Wenc,
                         const float* __restrict__ benc, float* __restrict__ pre) {
    __shared__ float row[ENC_];
    int b = blockIdx.x, t = blockIdx.y;
    const float* er = enc + ((size_t)b * T_ + t) * ENC_;
    for (int i = threadIdx.x; i < ENC_; i += 256) row[i] = er[i];
    __syncthreads();
    int a = threadIdx.x;
    const float* wr = Wenc + (size_t)a * ENC_;
    float s = 0.f;
    for (int d = 0; d < ENC_; ++d) s += row[d] * wr[d];
    pre[((size_t)b * T_ + t) * A_ + a] = s + benc[a];
}

__global__ void k_init(const int* __restrict__ enc_len, float* __restrict__ w,
                       unsigned short* __restrict__ x, unsigned short* __restrict__ xo,
                       float* __restrict__ dz, float* __restrict__ dc, int* __restrict__ prev) {
    int tid = blockIdx.x * blockDim.x + threadIdx.x;
    if (tid < B_) prev[tid] = 1;                       // BOS
    if (tid < B_ * H_) { dz[tid] = 0.f; dc[tid] = 0.f; }
    if (tid < B_ * KX_) x[tid] = 0;
    if (tid < B_ * KO_) xo[tid] = 0;
    if (tid < B_ * T_) {
        int b = tid / T_, t = tid % T_;
        int l = enc_len[b];
        w[tid] = (t < l) ? (1.0f / (float)l) : 0.0f;
    }
}

// ---------------- per-step kernels ----------------

// gather embedding row -> x[:, 0:512] (bf16)
__global__ void k_embed(const float* __restrict__ emb, const int* __restrict__ prev,
                        unsigned short* __restrict__ x) {
    int b = blockIdx.x, e = threadIdx.x;               // 256 threads, 2 elems each
    const float* r = emb + (size_t)prev[b] * E_;
    unsigned short* xr = x + (size_t)b * KX_;
    xr[e]       = f2bf(r[e]);
    xr[e + 256] = f2bf(r[e + 256]);
}

// One wave per 16-column tile, computing BOTH 16-row M-tiles (full M=32) so the
// B (weight) fragment is loaded once per k-step and reused for two WMMAs.
// A: [32, Ktot] row-major bf16 (lda). B: [N, Ktot] row-major bf16 (ldb) -> used as K x N.
__global__ void k_wmma_gemm(const unsigned short* __restrict__ A, int lda,
                            const unsigned short* __restrict__ Bm, int ldb,
                            const float* __restrict__ bias,
                            float* __restrict__ out, size_t ldo,
                            int Ktot) {
    int nT = blockIdx.x;
    int lane = threadIdx.x & 31;
    int mr = lane & 15, h = lane >> 4;
    const unsigned short* arow0 = A  + (size_t)mr * lda;         // M-tile 0
    const unsigned short* arow1 = A  + (size_t)(16 + mr) * lda;  // M-tile 1
    const unsigned short* brow  = Bm + (size_t)(nT * 16 + mr) * ldb;
    v8f acc0 = {}, acc1 = {};
    for (int kk = 0; kk < Ktot; kk += 32) {
        BfVec a0, a1, bb;
#pragma unroll
        for (int v = 0; v < 8; ++v) {
            // ISA 7.12.2 16-bit A layout: v<4 -> K=2v,2v+1 ; v>=4 -> K=16+2(v-4) ; +8 for hi lanes
            int kb = kk + ((v < 4) ? 2 * v : 16 + 2 * (v - 4)) + 8 * h;
            a0.u[v] = *(const unsigned int*)(arow0 + kb);   // pair (kb, kb+1) contiguous
            a1.u[v] = *(const unsigned int*)(arow1 + kb);
            bb.u[v] = *(const unsigned int*)(brow  + kb);
        }
        acc0 = __builtin_amdgcn_wmma_f32_16x16x32_bf16(false, a0.v, false, bb.v,
                                                       (short)0, acc0, false, false);
        acc1 = __builtin_amdgcn_wmma_f32_16x16x32_bf16(false, a1.v, false, bb.v,
                                                       (short)0, acc1, false, false);
    }
    int n = nT * 16 + mr;
    float bv = bias[n];
#pragma unroll
    for (int r = 0; r < 8; ++r) {
        int m = r + 8 * h;                 // D layout: VGPR r -> row r (+8 for hi lanes)
        out[(size_t)m * ldo + n]        = acc0[r] + bv;
        out[(size_t)(16 + m) * ldo + n] = acc1[r] + bv;
    }
}

__global__ void k_lstm(const float* __restrict__ gates, float* __restrict__ dc,
                       float* __restrict__ dz, unsigned short* __restrict__ x,
                       unsigned short* __restrict__ xo) {
    int idx = blockIdx.x * blockDim.x + threadIdx.x;   // 32*512
    int b = idx >> 9, hh = idx & 511;
    const float* g = gates + (size_t)b * G4_;
    float i = sigmf(g[hh]);
    float f = sigmf(g[512 + hh]);
    float gg = tanhf(g[1024 + hh]);
    float o = sigmf(g[1536 + hh]);
    float c = f * dc[idx] + i * gg;
    float z = o * tanhf(c);
    dc[idx] = c; dz[idx] = z;
    unsigned short zb = f2bf(z);
    x [(size_t)b * KX_ + 1024 + hh] = zb;              // lstm input for next step
    xo[(size_t)b * KO_ + hh]        = zb;              // logit input this step
}

// dec_t[b,a] = sum_h W_dec[a,h]*dec_z[b,h]
__global__ void k_dect(const float* __restrict__ Wdec, const float* __restrict__ dz,
                       float* __restrict__ dt) {
    __shared__ float z[H_];
    int b = blockIdx.x;
    for (int i = threadIdx.x; i < H_; i += 256) z[i] = dz[b * H_ + i];
    __syncthreads();
    int a = threadIdx.x;
    const float* wr = Wdec + (size_t)a * H_;
    float s = 0.f;
    for (int hh = 0; hh < H_; ++hh) s += wr[hh] * z[hh];
    dt[b * A_ + a] = s;
}

// fused location conv + tanh energy: one wave per (b,t)
__global__ void k_attn(const float* __restrict__ wprev, const float* __restrict__ convk,
                       const float* __restrict__ Watt, const float* __restrict__ pre,
                       const float* __restrict__ dect, const float* __restrict__ Wg,
                       float* __restrict__ e) {
    int b = blockIdx.x;
    int wave = threadIdx.x >> 5, lane = threadIdx.x & 31;
    int t = blockIdx.y * 8 + wave;
    if (t >= T_) return;
    float conv[C_];
#pragma unroll
    for (int c = 0; c < C_; ++c) conv[c] = 0.f;
    for (int k = lane; k < 2 * KK_ + 1; k += 32) {
        int tt = t + k - KK_;
        if (tt >= 0 && tt < T_) {
            float wv = wprev[b * T_ + tt];
#pragma unroll
            for (int c = 0; c < C_; ++c) conv[c] += convk[c * (2 * KK_ + 1) + k] * wv;
        }
    }
#pragma unroll
    for (int off = 16; off > 0; off >>= 1)
#pragma unroll
        for (int c = 0; c < C_; ++c) conv[c] += __shfl_xor(conv[c], off, 32);
    float ep = 0.f;
    for (int a = lane; a < A_; a += 32) {
        float ac = 0.f;
#pragma unroll
        for (int c = 0; c < C_; ++c) ac += conv[c] * Watt[a * C_ + c];
        float v = pre[((size_t)b * T_ + t) * A_ + a] + dect[b * A_ + a] + ac;
        ep += Wg[a] * tanhf(v);
    }
#pragma unroll
    for (int off = 16; off > 0; off >>= 1) ep += __shfl_xor(ep, off, 32);
    if (lane == 0) e[b * T_ + t] = ep;
}

__global__ void k_softmax(const float* __restrict__ e, float* __restrict__ w,
                          float* __restrict__ wsout, int step) {
    __shared__ float red[256];
    __shared__ float bmax, bsum;
    int b = blockIdx.x, tid = threadIdx.x;
    float m = -1e30f;
    for (int t = tid; t < T_; t += 256) m = fmaxf(m, 2.0f * e[b * T_ + t]);
    red[tid] = m; __syncthreads();
    for (int s = 128; s > 0; s >>= 1) { if (tid < s) red[tid] = fmaxf(red[tid], red[tid + s]); __syncthreads(); }
    if (tid == 0) bmax = red[0];
    __syncthreads();
    float sum = 0.f;
    for (int t = tid; t < T_; t += 256) sum += __expf(2.0f * e[b * T_ + t] - bmax);
    red[tid] = sum; __syncthreads();
    for (int s = 128; s > 0; s >>= 1) { if (tid < s) red[tid] += red[tid + s]; __syncthreads(); }
    if (tid == 0) bsum = red[0];
    __syncthreads();
    float inv = 1.0f / bsum;
    for (int t = tid; t < T_; t += 256) {
        float v = __expf(2.0f * e[b * T_ + t] - bmax) * inv;
        w[b * T_ + t] = v;
        wsout[((size_t)b * TD_ + step) * T_ + t] = v;
    }
}

// cvraw[b,d] = sum_t w[b,t]*enc[b,t,d]
__global__ void k_context(const float* __restrict__ w, const float* __restrict__ enc,
                          float* __restrict__ cvraw) {
    int b = blockIdx.x;
    int d = blockIdx.y * 256 + threadIdx.x;
    const float* eb = enc + (size_t)b * T_ * ENC_ + d;
    const float* wb = w + b * T_;
    float s = 0.f;
    for (int t = 0; t < T_; ++t) s += wb[t] * eb[(size_t)t * ENC_];
    cvraw[b * ENC_ + d] = s;
}

// c[b,j] = b_o[j] + sum_d cvraw[b,d]*W_o[j,d]  -> bf16 into x and xo
__global__ void k_cvec(const float* __restrict__ cvraw, const float* __restrict__ Wo,
                       const float* __restrict__ bo, unsigned short* __restrict__ x,
                       unsigned short* __restrict__ xo) {
    __shared__ float r[ENC_];
    int b = blockIdx.x;
    for (int i = threadIdx.x; i < ENC_; i += 512) r[i] = cvraw[b * ENC_ + i];
    __syncthreads();
    int j = threadIdx.x;                               // 512 threads
    const float* wr = Wo + (size_t)j * ENC_;
    float s = bo[j];
    for (int d = 0; d < ENC_; ++d) s += wr[d] * r[d];
    unsigned short cb = f2bf(s);
    x [(size_t)b * KX_ + 512 + j] = cb;
    xo[(size_t)b * KO_ + 512 + j] = cb;
}

// argmax + log-softmax-at-pred; pred feeds next step
__global__ void k_finalize(const float* __restrict__ lg, float* __restrict__ ys,
                           int* __restrict__ predout, int* __restrict__ prev, int step) {
    __shared__ float rm[256]; __shared__ int ri[256]; __shared__ float rs[256];
    __shared__ float bm;
    int b = blockIdx.x, tid = threadIdx.x;
    const float* row = lg + ((size_t)b * TD_ + step) * V_;
    float m = -1e30f; int mi = 0x7fffffff;
    for (int v = tid; v < V_; v += 256) {
        float xv = row[v];
        if (xv > m || (xv == m && v < mi)) { m = xv; mi = v; }
    }
    rm[tid] = m; ri[tid] = mi; __syncthreads();
    for (int s = 128; s > 0; s >>= 1) {
        if (tid < s) {
            float m2 = rm[tid + s]; int i2 = ri[tid + s];
            if (m2 > rm[tid] || (m2 == rm[tid] && i2 < ri[tid])) { rm[tid] = m2; ri[tid] = i2; }
        }
        __syncthreads();
    }
    if (tid == 0) bm = rm[0];
    __syncthreads();
    float sum = 0.f;
    for (int v = tid; v < V_; v += 256) sum += __expf(row[v] - bm);
    rs[tid] = sum; __syncthreads();
    for (int s = 128; s > 0; s >>= 1) { if (tid < s) rs[tid] += rs[tid + s]; __syncthreads(); }
    if (tid == 0) {
        ys[b * TD_ + step] = -logf(rs[0]);             // log_softmax at argmax = -log(sum exp(l-max))
        predout[b * TD_ + step] = ri[0];
        prev[b] = ri[0];
    }
}

// ---------------- host ----------------

extern "C" void kernel_launch(void* const* d_in, const int* in_sizes, int n_in,
                              void* d_out, int out_size, void* d_ws, size_t ws_size,
                              hipStream_t stream) {
    (void)in_sizes; (void)n_in; (void)out_size; (void)ws_size;
    const float* enc_pad  = (const float*)d_in[0];
    const int*   enc_len  = (const int*)  d_in[1];
    const float* embedding= (const float*)d_in[2];
    const float* W_ih     = (const float*)d_in[3];
    const float* b_ih     = (const float*)d_in[4];
    const float* W_hh     = (const float*)d_in[5];
    const float* b_hh     = (const float*)d_in[6];
    const float* W_enc    = (const float*)d_in[7];
    const float* b_enc    = (const float*)d_in[8];
    const float* W_dec    = (const float*)d_in[9];
    const float* W_att    = (const float*)d_in[10];
    const float* conv_k   = (const float*)d_in[11];
    const float* W_g      = (const float*)d_in[12];
    const float* W_o      = (const float*)d_in[13];
    const float* b_o      = (const float*)d_in[14];
    const float* W_out    = (const float*)d_in[15];
    const float* b_out    = (const float*)d_in[16];

    // workspace layout (bytes, 256-aligned)
    char* ws = (char*)d_ws;
    size_t off = 0;
    auto alloc = [&](size_t bytes) { size_t o = off; off = (off + bytes + 255) & ~(size_t)255; return o; };
    unsigned short* wcat  = (unsigned short*)(ws + alloc((size_t)G4_ * KX_ * 2));
    unsigned short* wout  = (unsigned short*)(ws + alloc((size_t)V_ * KO_ * 2));
    float* preenc = (float*)(ws + alloc((size_t)B_ * T_ * A_ * 4));
    float* gbias  = (float*)(ws + alloc((size_t)G4_ * 4));
    unsigned short* x  = (unsigned short*)(ws + alloc((size_t)B_ * KX_ * 2));
    unsigned short* xo = (unsigned short*)(ws + alloc((size_t)B_ * KO_ * 2));
    float* gates  = (float*)(ws + alloc((size_t)B_ * G4_ * 4));
    float* decz   = (float*)(ws + alloc((size_t)B_ * H_ * 4));
    float* decc   = (float*)(ws + alloc((size_t)B_ * H_ * 4));
    float* dect   = (float*)(ws + alloc((size_t)B_ * A_ * 4));
    float* ebuf   = (float*)(ws + alloc((size_t)B_ * T_ * 4));
    float* wbuf   = (float*)(ws + alloc((size_t)B_ * T_ * 4));
    float* cvraw  = (float*)(ws + alloc((size_t)B_ * ENC_ * 4));
    int*   prev   = (int*)  (ws + alloc((size_t)B_ * 4));

    // output layout: logits [B,TD,V] | ys [B,TD] | preds [B,TD] (int32 bits) | ws [B,TD,T]
    float* outf   = (float*)d_out;
    float* logits = outf;
    float* ys     = outf + (size_t)B_ * TD_ * V_;
    int*   preds  = (int*)(outf + (size_t)B_ * TD_ * V_ + (size_t)B_ * TD_);
    float* wsout  = outf + (size_t)B_ * TD_ * V_ + 2 * (size_t)B_ * TD_;

    // ---- one-time setup ----
    k_cvt_wcat<<<((size_t)G4_ * KX_ + 255) / 256, 256, 0, stream>>>(W_ih, W_hh, wcat);
    k_cvt_wout<<<((size_t)V_ * KO_ + 255) / 256, 256, 0, stream>>>(W_out, wout);
    k_bias<<<(G4_ + 255) / 256, 256, 0, stream>>>(b_ih, b_hh, gbias);
    k_preenc<<<dim3(B_, T_), 256, 0, stream>>>(enc_pad, W_enc, b_enc, preenc);
    k_init<<<(B_ * KX_ + 255) / 256, 256, 0, stream>>>(enc_len, wbuf, x, xo, decz, decc, prev);

    // ---- sequential decode ----
    for (int step = 0; step < TD_; ++step) {
        k_embed<<<B_, 256, 0, stream>>>(embedding, prev, x);
        // gates = [emb|c|dec_z] @ [W_ih|W_hh]^T + (b_ih+b_hh)   (WMMA bf16, full M per wave)
        k_wmma_gemm<<<G4_ / 16, 32, 0, stream>>>(x, KX_, wcat, KX_, gbias,
                                                 gates, (size_t)G4_, KX_);
        k_lstm<<<(B_ * H_) / 256, 256, 0, stream>>>(gates, decc, decz, x, xo);
        k_dect<<<B_, 256, 0, stream>>>(W_dec, decz, dect);
        k_attn<<<dim3(B_, (T_ + 7) / 8), 256, 0, stream>>>(wbuf, conv_k, W_att, preenc, dect, W_g, ebuf);
        k_softmax<<<B_, 256, 0, stream>>>(ebuf, wbuf, wsout, step);
        k_context<<<dim3(B_, ENC_ / 256), 256, 0, stream>>>(wbuf, enc_pad, cvraw);
        k_cvec<<<B_, 512, 0, stream>>>(cvraw, W_o, b_o, x, xo);
        // logits = [dec_z|c] @ W_out^T + b_out   (WMMA bf16), written into d_out at [:,step,:]
        k_wmma_gemm<<<V_ / 16, 32, 0, stream>>>(xo, KO_, wout, KO_, b_out,
                                                logits + (size_t)step * V_,
                                                (size_t)TD_ * V_, KO_);
        k_finalize<<<B_, 256, 0, stream>>>(logits, ys, preds, prev, step);
    }
}
